// AttentionLayer_22187801051597
// MI455X (gfx1250) — matrix operand order repaired and checked by reading, not compiled, not run on hardware
//
#include <hip/hip_runtime.h>
#include <hip/hip_bf16.h>
#include <math.h>

typedef __attribute__((ext_vector_type(16))) __bf16 v16bf;
typedef __attribute__((ext_vector_type(8)))  float  v8f;

#define B_  4
#define T_  2048
#define DIN 768
#define DK  128
#define DV  768
#define BT  (B_*T_)
#define KT_STRIDE 136   // LDS K-tile row stride in bf16 (128 + 8 pad -> spreads banks)

// ---------------------------------------------------------------------------
// WMMA fragment loaders, wave32; contiguous k order k = 16*(lane/16) + s for both A and B.
// Both loaders read 16 CONTIGUOUS bf16 (32 B) per lane -> 2x b128 loads.
// ---------------------------------------------------------------------------
static __device__ __forceinline__ v16bf load_a_frag(const __bf16* __restrict__ src, int lda) {
  const int lane = threadIdx.x & 31;
  const int row  = lane & 15;
  const int kb   = (lane >> 4) << 4;   // 0 or 16
  const __bf16* p = src + (size_t)row * lda;
  v16bf a;
#pragma unroll
  for (int i = 0; i < 8; ++i) a[i]     = p[kb + i];
#pragma unroll
  for (int i = 0; i < 8; ++i) a[8 + i] = p[kb + 8 + i];
  return a;
}

// B[k][n] = src[n*ld + k]  (transposed operand: lane owns column n)
static __device__ __forceinline__ v16bf load_bT_frag(const __bf16* __restrict__ src, int ld) {
  const int lane = threadIdx.x & 31;
  const int col  = lane & 15;
  const int kb   = (lane >> 4) << 4;   // 0 or 16
  const __bf16* p = src + (size_t)col * ld + kb;
  v16bf b;
#pragma unroll
  for (int i = 0; i < 16; ++i) b[i] = p[i];
  return b;
}

static __device__ __forceinline__ v8f wmma_bf16(v16bf a, v16bf b, v8f c) {
  return __builtin_amdgcn_wmma_f32_16x16x32_bf16(false, a, false, b, (short)0, c, false, false);
}

// ---------------------------------------------------------------------------
// CDNA5 async copy: 16 B global -> LDS, tracked by ASYNCcnt
// ---------------------------------------------------------------------------
static __device__ __forceinline__ void async_copy_16B(unsigned lds_off, const void* gaddr) {
  asm volatile("global_load_async_to_lds_b128 %0, %1, off"
               :: "v"(lds_off), "v"(gaddr) : "memory");
}

// Stage one 32x128 bf16 K tile (8 KB) into LDS: 256 threads x 2 chunks of 16 B.
static __device__ __forceinline__ void stage_k_tile(const __bf16* __restrict__ gK,
                                                    __bf16* lbuf, int tid) {
  const unsigned lbase = (unsigned)(uintptr_t)lbuf;   // low 32 bits == LDS offset
#pragma unroll
  for (int q = 0; q < 2; ++q) {
    const int c   = tid * 2 + q;   // 512 chunks of 16 B
    const int row = c >> 4;        // 32 key rows
    const int sub = c & 15;        // 16 x 16 B per 256 B row
    async_copy_16B(lbase + (unsigned)(row * (KT_STRIDE * 2) + sub * 16),
                   (const char*)gK + row * (DK * 2) + sub * 16);
  }
}

// ---------------------------------------------------------------------------
// f32 -> bf16, 8 elements per thread (b128 load/store)
// ---------------------------------------------------------------------------
__global__ void to_bf16_kernel(const float* __restrict__ in, __bf16* __restrict__ out, int n8) {
  int i = blockIdx.x * blockDim.x + threadIdx.x;
  const int stride = gridDim.x * blockDim.x;
  for (; i < n8; i += stride) {
#pragma unroll
    for (int j = 0; j < 8; ++j) out[8 * i + j] = (__bf16)in[8 * i + j];
  }
}

// f32 [K][N] -> bf16 transposed [N][K]; consecutive threads walk k (coalesced stores)
__global__ void tconv_kernel(const float* __restrict__ in, __bf16* __restrict__ out,
                             int K, int N) {
  int idx = blockIdx.x * blockDim.x + threadIdx.x;
  if (idx >= K * N) return;
  const int k = idx % K;
  const int n = idx / K;
  out[(size_t)n * K + k] = (__bf16)in[(size_t)k * N + n];
}

// ---------------------------------------------------------------------------
// Projection GEMM:  C[m,n] = (A[m,:] @ Wt[n,:] + bias[n]) * scale  -> bf16
// tout=0: Out[m*N+n].  tout=1: Out[n*BT+m] (V^T).
// ---------------------------------------------------------------------------
__global__ __launch_bounds__(256) void qkv_gemm_kernel(
    const __bf16* __restrict__ A, const __bf16* __restrict__ Wt,
    const float* __restrict__ bias, __bf16* __restrict__ Out,
    int N, float scale, int tout) {
  const int w     = threadIdx.x >> 5;
  const int lane  = threadIdx.x & 31;
  const int mbase = blockIdx.x * 128 + w * 16;
  const int nbase = blockIdx.y * 64;

  v8f acc[4] = {};
  const __bf16* arow = A + (size_t)mbase * DIN;

  for (int kk = 0; kk < DIN; kk += 32) {
    if (kk + 64 < DIN)
      __builtin_prefetch(arow + kk + 64, 0, 1);   // global_prefetch_b8
    // gather ALL fragments first -> one load clause, then 4 back-to-back WMMAs
    v16bf bf[4];
#pragma unroll
    for (int nt = 0; nt < 4; ++nt)
      bf[nt] = load_bT_frag(Wt + (size_t)(nbase + nt * 16) * DIN + kk, DIN);
    v16bf af = load_a_frag(arow + kk, DIN);
#pragma unroll
    for (int nt = 0; nt < 4; ++nt)
      acc[nt] = wmma_bf16(af, bf[nt], acc[nt]);
  }

  const int rrow = (lane >> 4) * 8;   // C layout: row = r + 8*(lane>=16)
  const int coll = lane & 15;
  if (tout) {
#pragma unroll
    for (int nt = 0; nt < 4; ++nt) {
      const int col = nbase + nt * 16 + coll;
      const float bb = bias[col];
#pragma unroll
      for (int r = 0; r < 8; ++r)
        Out[(size_t)col * BT + (mbase + rrow + r)] = (__bf16)((acc[nt][r] + bb) * scale);
    }
  } else {
#pragma unroll
    for (int nt = 0; nt < 4; ++nt) {
      const int col = nbase + nt * 16 + coll;
      const float bb = bias[col];
#pragma unroll
      for (int r = 0; r < 8; ++r)
        Out[(size_t)(mbase + rrow + r) * N + col] = (__bf16)((acc[nt][r] + bb) * scale);
    }
  }
}

// ---------------------------------------------------------------------------
// Flash attention. Grid: (T/32, B). Block: 256 threads = 8 waves.
// Block owns 32 query rows; wave w: q-subtile (w>>2), d_v slice 192*(w&3).
// K tiles double-buffered in LDS via global_load_async_to_lds_b128.
// Q carries 1/sqrt(dk). V transposed [DV][BT].
// ---------------------------------------------------------------------------
__global__ __launch_bounds__(256) void attn_kernel(
    const __bf16* __restrict__ Q, const __bf16* __restrict__ K,
    const __bf16* __restrict__ Vt, float* __restrict__ Out) {
  __shared__ __bf16 kbuf[2][32 * KT_STRIDE];   // 2 x 8.5 KB K tiles
  __shared__ __bf16 plds[8][16 * 32];          // per-wave P staging

  const int tid   = threadIdx.x;
  const int w     = tid >> 5;
  const int lane  = tid & 31;
  const int b     = blockIdx.y;
  const int qbase = blockIdx.x * 32 + (w >> 2) * 16;
  const int dvb   = (w & 3) * 192;
  const size_t qrow0 = (size_t)b * T_ + qbase;
  const __bf16* Krow0 = K + (size_t)b * T_ * DK;

  // Q fragments: 16 rows x 128 (4 chunks of 32)
  v16bf qf[4];
#pragma unroll
  for (int kk = 0; kk < 4; ++kk)
    qf[kk] = load_a_frag(Q + qrow0 * DK + kk * 32, DK);

  v8f o[12] = {};
  float m[8], l[8];
#pragma unroll
  for (int r = 0; r < 8; ++r) { m[r] = -INFINITY; l[r] = 0.0f; }

  __bf16* pl = &plds[w][0];
  const int rrow = (lane >> 4) * 8;
  const int colk = lane & 15;
  const __bf16* Vbase = Vt + (size_t)b * T_;   // + n*BT + t

  // preload first K tile
  stage_k_tile(Krow0, &kbuf[0][0], tid);

  for (int j = 0; j < T_; j += 32) {
    const int cur = (j >> 5) & 1;
    // my async copies done -> barrier -> whole tile visible to all waves
    asm volatile("s_wait_asynccnt 0" ::: "memory");
    __syncthreads();
    if (j + 32 < T_)
      stage_k_tile(Krow0 + (size_t)(j + 32) * DK, &kbuf[cur ^ 1][0], tid);

    // S tiles from LDS: keys [j,j+16) and [j+16,j+32); 2 x ds_load_b128 per frag
    const __bf16* Kl = &kbuf[cur][0];
    v8f s0 = {}, s1 = {};
#pragma unroll
    for (int kk = 0; kk < 4; ++kk) {
      s0 = wmma_bf16(qf[kk], load_bT_frag(Kl + kk * 32, KT_STRIDE), s0);
      s1 = wmma_bf16(qf[kk], load_bT_frag(Kl + 16 * KT_STRIDE + kk * 32, KT_STRIDE), s1);
    }

    // online softmax update, per row r (row replicated across 16 lanes)
#pragma unroll
    for (int r = 0; r < 8; ++r) {
      float v0 = s0[r], v1 = s1[r];
      float mx = fmaxf(v0, v1);
#pragma unroll
      for (int msk = 1; msk < 16; msk <<= 1) mx = fmaxf(mx, __shfl_xor(mx, msk, 32));
      const float mnew = fmaxf(m[r], mx);
      const float corr = __expf(m[r] - mnew);
      const float p0 = __expf(v0 - mnew);
      const float p1 = __expf(v1 - mnew);
      float ps = p0 + p1;
#pragma unroll
      for (int msk = 1; msk < 16; msk <<= 1) ps += __shfl_xor(ps, msk, 32);
      l[r] = l[r] * corr + ps;
      m[r] = mnew;
#pragma unroll
      for (int nt = 0; nt < 12; ++nt) o[nt][r] *= corr;
      const int prow = rrow + r;
      pl[prow * 32 + colk]      = (__bf16)p0;
      pl[prow * 32 + 16 + colk] = (__bf16)p1;
    }

    // within-wave LDS store->load ordering for the P relayout
    asm volatile("s_wait_dscnt 0" ::: "memory");

    v16bf pf = load_a_frag(pl, 32);              // 2 x ds_load_b128
    const __bf16* Vp = Vbase + j;                // V^T: row n, contiguous keys
#pragma unroll
    for (int nt = 0; nt < 12; ++nt)
      o[nt] = wmma_bf16(pf, load_bT_frag(Vp + (size_t)(dvb + nt * 16) * BT, BT), o[nt]);
  }

  // epilogue: divide by row sum, write f32 output [B,T,DV]
#pragma unroll
  for (int r = 0; r < 8; ++r) {
    const float inv = 1.0f / l[r];
    const size_t orow = (qrow0 + rrow + r) * DV + dvb;
#pragma unroll
    for (int nt = 0; nt < 12; ++nt)
      Out[orow + nt * 16 + colk] = o[nt][r] * inv;
  }
}

// ---------------------------------------------------------------------------
// Host launch
// ---------------------------------------------------------------------------
extern "C" void kernel_launch(void* const* d_in, const int* in_sizes, int n_in,
                              void* d_out, int out_size, void* d_ws, size_t ws_size,
                              hipStream_t stream) {
  const float* x  = (const float*)d_in[0];
  const float* Wq = (const float*)d_in[1];
  const float* bq = (const float*)d_in[2];
  const float* Wk = (const float*)d_in[3];
  const float* bk = (const float*)d_in[4];
  const float* Wv = (const float*)d_in[5];
  const float* bv = (const float*)d_in[6];
  float* out = (float*)d_out;

  char* ws = (char*)d_ws;
  __bf16* xb  = (__bf16*)ws;  ws += (size_t)BT  * DIN * 2;   // x bf16 [BT][DIN]
  __bf16* wqt = (__bf16*)ws;  ws += (size_t)DIN * DK  * 2;   // Wq^T [DK][DIN]
  __bf16* wkt = (__bf16*)ws;  ws += (size_t)DIN * DK  * 2;   // Wk^T [DK][DIN]
  __bf16* wvt = (__bf16*)ws;  ws += (size_t)DIN * DV  * 2;   // Wv^T [DV][DIN]
  __bf16* qb  = (__bf16*)ws;  ws += (size_t)BT  * DK  * 2;   // Q bf16 (scaled)
  __bf16* kb  = (__bf16*)ws;  ws += (size_t)BT  * DK  * 2;   // K bf16
  __bf16* vt  = (__bf16*)ws;  ws += (size_t)BT  * DV  * 2;   // V^T bf16 [DV][BT]

  to_bf16_kernel<<<1024, 256, 0, stream>>>(x, xb, (BT * DIN) / 8);
  tconv_kernel<<<(DIN * DK + 255) / 256, 256, 0, stream>>>(Wq, wqt, DIN, DK);
  tconv_kernel<<<(DIN * DK + 255) / 256, 256, 0, stream>>>(Wk, wkt, DIN, DK);
  tconv_kernel<<<(DIN * DV + 255) / 256, 256, 0, stream>>>(Wv, wvt, DIN, DV);

  const float scale = 0.08838834764831845f;  // 1/sqrt(128), folded into Q
  qkv_gemm_kernel<<<dim3(BT / 128, DK / 64), 256, 0, stream>>>(xb, wqt, bq, qb, DK, scale, 0);
  qkv_gemm_kernel<<<dim3(BT / 128, DK / 64), 256, 0, stream>>>(xb, wkt, bk, kb, DK, 1.0f, 0);
  qkv_gemm_kernel<<<dim3(BT / 128, DV / 64), 256, 0, stream>>>(xb, wvt, bv, vt, DV, 1.0f, 1);

  attn_kernel<<<dim3(T_ / 32, B_), 256, 0, stream>>>(qb, kb, vt, out);
}
